// Attention_62938450756123
// MI455X (gfx1250) — matrix operand order repaired and checked by reading, not compiled, not run on hardware
//
#include <hip/hip_runtime.h>

typedef _Float16 f16;
typedef __attribute__((ext_vector_type(16))) _Float16 v16h;
typedef __attribute__((ext_vector_type(8)))  _Float16 v8h;
typedef __attribute__((ext_vector_type(4)))  _Float16 v4h;
typedef __attribute__((ext_vector_type(8)))  float    v8f;
typedef __attribute__((ext_vector_type(4)))  float    v4f;

#define SL 2048
#define TL 2048
#define BS 16
#define H  512
#define NWAVE 8

union V16U { v16h v; v8h h[2]; };

// Build a 16-half WMMA fragment from two 16B-aligned 8-half chunks.
__device__ __forceinline__ v16h make_frag(const f16* lo, const f16* hi) {
    V16U u;
    u.h[0] = *(const v8h*)lo;
    u.h[1] = *(const v8h*)hi;
    return u.v;
}

// B fragments for S = Q.K^T : 32(h) x 16(s), K(h) contiguous per lane in oe16.
__device__ __forceinline__ void load_bk(const f16* __restrict__ oe16, int b, int s0,
                                        int hs, int l16, int lh, v16h out[2][2]) {
    for (int kc = 0; kc < 2; ++kc)
        for (int sh = 0; sh < 2; ++sh) {
            const int s  = s0 + sh * 16 + l16;
            const int hb = hs + kc * 32 + lh * 16;
            const f16* base = oe16 + ((size_t)(b * SL + s)) * H + hb;
            out[kc][sh] = make_frag(base, base + 8);
        }
}

// B fragments for ctx += P.V : 32(s) x 16(h), K(s) contiguous per lane in oeT16.
__device__ __forceinline__ void load_bv(const f16* __restrict__ oeT16, int b, int s0,
                                        int hs, int l16, int lh, v16h out[4]) {
    for (int nc = 0; nc < 4; ++nc) {
        const int h = hs + nc * 16 + l16;
        const f16* base = oeT16 + ((size_t)(b * H + h)) * SL + s0 + lh * 16;
        out[nc] = make_frag(base, base + 8);
    }
}

// ---------------------------------------------------------------------------
// Prep 1: od16[b][t][h] = (f16) out_d[t][b][h]   (coalesced both sides)
// ---------------------------------------------------------------------------
__global__ __launch_bounds__(256) void prep_od(const float* __restrict__ out_d,
                                               f16* __restrict__ od16) {
    const size_t idx = (size_t)blockIdx.x * 256 + threadIdx.x; // = (b*TL+t)*H+h
    const int h = (int)(idx % H);
    const int t = (int)((idx / H) % TL);
    const int b = (int)(idx / ((size_t)H * TL));
    od16[idx] = (f16)out_d[((size_t)t * BS + b) * H + h];
}

// ---------------------------------------------------------------------------
// Prep 2: direction-sum + dual-layout store.
//   oe16 [b][s][h] = (f16)(out_e[s][b][h] + out_e[s][b][h+H])
//   oeT16[b][h][s] = same, transposed (LDS-tiled so both stores coalesce)
// ---------------------------------------------------------------------------
__global__ __launch_bounds__(256) void prep_oe(const float* __restrict__ out_e,
                                               f16* __restrict__ oe16,
                                               f16* __restrict__ oeT16) {
    __shared__ f16 tile[32][33];
    const int tx = threadIdx.x & 31;
    const int ty = threadIdx.x >> 5;          // 0..7
    const int h0 = blockIdx.x * 32;
    const int s0 = blockIdx.y * 32;
    const int b  = blockIdx.z;
    for (int i = 0; i < 4; ++i) {
        const int sy = ty + i * 8;
        const int s  = s0 + sy;
        const int h  = h0 + tx;
        const float* p = out_e + ((size_t)s * BS + b) * (2 * H);
        const f16 hv = (f16)(p[h] + p[h + H]);
        oe16[((size_t)b * SL + s) * H + h] = hv;
        tile[sy][tx] = hv;
    }
    __syncthreads();
    for (int i = 0; i < 4; ++i) {
        const int hy = ty + i * 8;
        const int h  = h0 + hy;
        const int s  = s0 + tx;
        oeT16[((size_t)b * H + h) * SL + s] = tile[tx][hy];
    }
}

// ---------------------------------------------------------------------------
// Streaming attention (faithful naive softmax over s: exp / sum, no max).
// Grid: (TL/32, BS). Block: 256 (8 waves). Wave w owns h-slice [64w, 64w+64).
// Register-double-buffered K fragments + early-issued V fragments so global
// L2 latency overlaps an entire iteration of WMMA + barriers + exp phase.
// ---------------------------------------------------------------------------
__global__ __launch_bounds__(256) void attn_fa(const f16* __restrict__ od16,
                                               const f16* __restrict__ oe16,
                                               const f16* __restrict__ oeT16,
                                               float* __restrict__ out) {
    __shared__ __align__(16) float s_part[NWAVE][2][16][32]; // 32 KB split-K partials
    __shared__ __align__(16) f16   s_p[2][16][32];           // P tile (f16)
    __shared__ float s_rowsum[32];
    __shared__ float s_rowinv[32];

    const int tid  = threadIdx.x;
    const int lane = tid & 31;
    const int wave = tid >> 5;
    const int l16  = lane & 15;
    const int lh   = lane >> 4;              // 0 | 1 (lane half)
    const int b    = blockIdx.y;
    const int t0   = blockIdx.x * 32;
    const int hs   = wave * 64;              // this wave's h-slice base

    // Q fragments (A: 16t x 32h) resident for the whole s loop.
    // A layout: lane row M=l16; K chunk base = lh*16; contiguous [+0..15]
    v16h aq[2][2];
    for (int m = 0; m < 2; ++m)
        for (int kc = 0; kc < 2; ++kc) {
            const int t  = t0 + m * 16 + l16;
            const int hb = hs + kc * 32 + lh * 16;
            const f16* base = od16 + ((size_t)(b * TL + t)) * H + hb;
            aq[m][kc] = make_frag(base, base + 8);
        }

    v8f c[2][4];
    {
        v8f z = {};
        for (int m = 0; m < 2; ++m)
            for (int n = 0; n < 4; ++n) c[m][n] = z;
    }

    if (tid < 32) s_rowsum[tid] = 0.0f;

    // Prime the K-fragment double buffer for s0 = 0.
    v16h bkbuf[2][2][2];                     // [parity][kc][shalf]
    load_bk(oe16, b, 0, hs, l16, lh, bkbuf[0]);

    __syncthreads();

#pragma unroll 2
    for (int it = 0; it < SL / 32; ++it) {
        const int s0 = it * 32;
        const int p  = it & 1;
        const int pn = p ^ 1;
        const int sn = (s0 + 32 < SL) ? (s0 + 32) : 0; // wrap: harmless dead load

        // Issue next-step K loads and this-step V loads NOW; consume much later.
        load_bk(oe16, b, sn, hs, l16, lh, bkbuf[pn]);
        v16h bv[4];
        load_bv(oeT16, b, s0, hs, l16, lh, bv);

        // ---- Phase 1: partial S = Q . K^T over this wave's 64-wide h slice
        for (int m = 0; m < 2; ++m)
            for (int sh = 0; sh < 2; ++sh) {
                v8f acc = {};
                acc = __builtin_amdgcn_wmma_f32_16x16x32_f16(
                        false, aq[m][0], false, bkbuf[p][0][sh], (short)0, acc, false, false);
                acc = __builtin_amdgcn_wmma_f32_16x16x32_f16(
                        false, aq[m][1], false, bkbuf[p][1][sh], (short)0, acc, false, false);
                // C layout: VGPR r -> row M = r + 8*lh, col N = l16
                for (int r = 0; r < 8; ++r)
                    s_part[wave][m][r + lh * 8][sh * 16 + l16] = acc[r];
            }
        __syncthreads();

        // ---- Phase 2: cross-wave reduce, exp (naive, no max), rowsum, P->f16
        {
            const int base = tid * 4;            // 1024 elems, 4/thread, same row
            const int m = base >> 9;
            const int t = (base >> 5) & 15;
            const int s = base & 31;
            v4f acc = *(const v4f*)&s_part[0][m][t][s];
            for (int w = 1; w < NWAVE; ++w)
                acc += *(const v4f*)&s_part[w][m][t][s];
            v4h ph;
            float psum = 0.0f;
            for (int j = 0; j < 4; ++j) {
                const float pe = __expf(acc[j]);
                ph[j] = (f16)pe;
                psum += pe;
            }
            *(v4h*)&s_p[m][t][s] = ph;
            atomicAdd(&s_rowsum[m * 16 + t], psum);   // ds_add_f32
        }
        __syncthreads();

        // ---- Phase 3: context += P . V on this wave's h slice (N-split)
        v16h ap[2];
        for (int m = 0; m < 2; ++m) {
            const f16* base = &s_p[m][l16][lh * 16];   // A layout over s (K-dim)
            ap[m] = make_frag(base, base + 8);
        }
        for (int nc = 0; nc < 4; ++nc)
            for (int m = 0; m < 2; ++m)
                c[m][nc] = __builtin_amdgcn_wmma_f32_16x16x32_f16(
                        false, ap[m], false, bv[nc], (short)0, c[m][nc], false, false);
        __syncthreads(); // protect s_part / s_p for next iteration
    }

    // ---- Epilogue: divide by rowsum, write [tl, bs, h] f32
    if (tid < 32) s_rowinv[tid] = 1.0f / s_rowsum[tid];
    __syncthreads();

    for (int m = 0; m < 2; ++m)
        for (int r = 0; r < 8; ++r) {
            const int trow = m * 16 + r + lh * 8;
            const float inv = s_rowinv[trow];
            const int t = t0 + trow;
            for (int nc = 0; nc < 4; ++nc) {
                const int h = hs + nc * 16 + l16;
                out[((size_t)t * BS + b) * H + h] = c[m][nc][r] * inv;
            }
        }
}

// ---------------------------------------------------------------------------
extern "C" void kernel_launch(void* const* d_in, const int* in_sizes, int n_in,
                              void* d_out, int out_size, void* d_ws, size_t ws_size,
                              hipStream_t stream) {
    // d_in[0] = in_e (int64, unused by forward)
    const float* out_e = (const float*)d_in[1];   // [sl, bs, 2h] f32
    const float* out_d = (const float*)d_in[2];   // [tl, bs, h]  f32
    float* out = (float*)d_out;                   // [tl, bs, h]  f32

    const size_t N = (size_t)BS * SL * H;         // 16M elems per f16 tensor
    f16* od16  = (f16*)d_ws;                      // 32 MB
    f16* oe16  = od16 + N;                        // 32 MB
    f16* oeT16 = oe16 + N;                        // 32 MB (96 MB total: L2-resident)

    prep_od<<<dim3((unsigned)(N / 256)), 256, 0, stream>>>(out_d, od16);
    prep_oe<<<dim3(H / 32, SL / 32, BS), 256, 0, stream>>>(out_e, oe16, oeT16);
    attn_fa<<<dim3(TL / 32, BS), 256, 0, stream>>>(od16, oe16, oeT16, out);
}
